// MultiHeadSelfAttention_14113262534988
// MI455X (gfx1250) — compile-verified, matrix operand order checked by reading, not run on hardware
//
#include <hip/hip_runtime.h>
#include <hip/hip_bf16.h>

// ---------------------------------------------------------------------------
// MHSA forward for MI455X (gfx1250), bf16 WMMA path, fp32 accumulate.
// B=2, N=2048, C=1024, H=16, HD=64.  SCALE = 1/8 folded into Q.
// GEMMs are register-blocked: one wave owns a (MT*16)x(NT*16) macro-tile,
// giving MT*NT WMMAs per MT+NT fragment loads (32 flops/byte at 4x4).
// ---------------------------------------------------------------------------

typedef __attribute__((ext_vector_type(16))) __bf16 v16bf;
typedef __attribute__((ext_vector_type(8)))  float  v8f;

#define BN      4096          // B*N rows
#define CIN     1024
#define C3      3072
#define SEQ     2048
#define HD      64
#define NHEAD   16
#define NBH     32            // B*H

__device__ __forceinline__ v8f wmma_bf16(v16bf a, v16bf b, v8f c) {
  return __builtin_amdgcn_wmma_f32_16x16x32_bf16(false, a, false, b,
                                                 (short)0, c, false, false);
}

// Fragment loader: elements 0..7 at p[0..7], 8..15 at p[16..23]
// (ISA 7.12.2: 16-bit A/B 16x32 layout, K = g*16 + half*8 + e).
__device__ __forceinline__ v16bf load_frag(const __hip_bfloat16* p) {
  union { v16bf v; uint4 q[2]; } u;
  u.q[0] = *(const uint4*)(p);
  u.q[1] = *(const uint4*)(p + 16);
  return u.v;
}

// -------------------- stage 1: fp32 -> bf16 conversions ---------------------

__global__ void cvt_f32_bf16(const float* __restrict__ in,
                             __hip_bfloat16* __restrict__ out, int n) {
  int i = blockIdx.x * blockDim.x + threadIdx.x;
  if (i < n) out[i] = __float2bfloat16(in[i]);
}

// in: [rows][cols] row-major fp32  ->  out: [cols][rows] bf16
__global__ void transpose_cvt(const float* __restrict__ in,
                              __hip_bfloat16* __restrict__ out,
                              int rows, int cols) {
  int i = blockIdx.x * blockDim.x + threadIdx.x;
  if (i < rows * cols) {
    int r = i / cols, c = i % cols;
    out[(size_t)c * rows + r] = __float2bfloat16(in[i]);
  }
}

// -------------------- register-blocked wave GEMM ---------------------------
// A: [M][K] row-major bf16, Bt: [N][K] row-major bf16 (B transposed).
// Wave computes a (MT*16)x(NT*16) macro-tile at (m0, n0).
template <int MT, int NT>
__device__ __forceinline__ void wave_gemm_block(const __hip_bfloat16* A,
                                                const __hip_bfloat16* Bt,
                                                int lda, int ldb,
                                                int m0, int n0, int K,
                                                v8f acc[MT][NT]) {
  const int lane = threadIdx.x & 31;
  const int r = lane & 15, hw = lane >> 4;
  const __hip_bfloat16* ap[MT];
  const __hip_bfloat16* bp[NT];
#pragma unroll
  for (int mi = 0; mi < MT; ++mi)
    ap[mi] = A + (size_t)(m0 + mi * 16 + r) * lda + hw * 8;
#pragma unroll
  for (int ni = 0; ni < NT; ++ni)
    bp[ni] = Bt + (size_t)(n0 + ni * 16 + r) * ldb + hw * 8;

  for (int k = 0; k < K; k += 32) {
    v16bf af[MT], bf[NT];
#pragma unroll
    for (int mi = 0; mi < MT; ++mi) af[mi] = load_frag(ap[mi] + k);
#pragma unroll
    for (int ni = 0; ni < NT; ++ni) bf[ni] = load_frag(bp[ni] + k);
#pragma unroll
    for (int mi = 0; mi < MT; ++mi)
#pragma unroll
      for (int ni = 0; ni < NT; ++ni)
        acc[mi][ni] = wmma_bf16(af[mi], bf[ni], acc[mi][ni]);
  }
}

// -------------------- stage 2: QKV projection ------------------------------
// 64x64 macro-tile per wave. out col in [0,3072): which = col/1024 (q,k,v);
// head = (col%1024)/64; d = col%64.  Q,K stored [bh][n][d] (Q pre-scaled by
// 0.125); V stored transposed [bh][d][n].
__global__ __launch_bounds__(256)
void qkv_gemm_kernel(const __hip_bfloat16* __restrict__ xb,
                     const __hip_bfloat16* __restrict__ wT,
                     const float* __restrict__ bias,
                     __hip_bfloat16* __restrict__ Qb,
                     __hip_bfloat16* __restrict__ Kb,
                     __hip_bfloat16* __restrict__ Vt) {
  const int wave = (blockIdx.x * blockDim.x + threadIdx.x) >> 5;
  const int tn = wave % (C3 / 64);
  const int tm = wave / (C3 / 64);
  const int m0 = tm * 64, n0 = tn * 64;

  v8f acc[4][4];
#pragma unroll
  for (int mi = 0; mi < 4; ++mi)
#pragma unroll
    for (int ni = 0; ni < 4; ++ni) acc[mi][ni] = (v8f){};
  wave_gemm_block<4, 4>(xb, wT, CIN, CIN, m0, n0, CIN, acc);

  const int lane = threadIdx.x & 31;
  const int r = lane & 15, hw = lane >> 4;
#pragma unroll
  for (int ni = 0; ni < 4; ++ni) {
    const int col = n0 + ni * 16 + r;
    const float bv = bias[col];
    const int which = col >> 10;
    const int c = col & 1023;
    const int head = c >> 6, d = c & 63;
#pragma unroll
    for (int mi = 0; mi < 4; ++mi) {
#pragma unroll
      for (int j = 0; j < 8; ++j) {
        const int m = m0 + mi * 16 + hw * 8 + j;
        const int bb = m >> 11, nrow = m & 2047;
        const float val = acc[mi][ni][j] + bv;
        const int bh = bb * NHEAD + head;
        if (which == 0)
          Qb[((size_t)bh * SEQ + nrow) * HD + d] = __float2bfloat16(val * 0.125f);
        else if (which == 1)
          Kb[((size_t)bh * SEQ + nrow) * HD + d] = __float2bfloat16(val);
        else
          Vt[((size_t)bh * HD + d) * SEQ + nrow] = __float2bfloat16(val);
      }
    }
  }
}

// -------------------- stage 3: flash attention -----------------------------
// One wave = one 16-row query block of one (b,h). Streams 32 keys/iteration:
// 4 WMMAs for S = Q@K^T, online softmax, 4 WMMAs for O += P@V.
__global__ __launch_bounds__(256)
void attn_kernel(const __hip_bfloat16* __restrict__ Qb,
                 const __hip_bfloat16* __restrict__ Kb,
                 const __hip_bfloat16* __restrict__ Vt,
                 __hip_bfloat16* __restrict__ Aout) {
  __shared__ alignas(16) __hip_bfloat16 stage[8][16 * 32];  // per-wave P tile

  const int warp = threadIdx.x >> 5;
  const int lane = threadIdx.x & 31;
  const int r = lane & 15, hw = lane >> 4;

  const int task = blockIdx.x * 8 + warp;    // 32 bh * 128 qblocks = 4096
  const int bh = task >> 7;
  const int qb = task & 127;
  const int b = bh >> 4, head = bh & 15;

  const __hip_bfloat16* Qp = Qb + (size_t)bh * SEQ * HD;
  const __hip_bfloat16* Kp = Kb + (size_t)bh * SEQ * HD;
  const __hip_bfloat16* Vp = Vt + (size_t)bh * HD * SEQ;

  // Q block fragments: 16x64 = two 16x32 A-frags (scale already folded in)
  const __hip_bfloat16* qrow = Qp + (size_t)(qb * 16 + r) * HD + hw * 8;
  const v16bf aq0 = load_frag(qrow);
  const v16bf aq1 = load_frag(qrow + 32);

  v8f o0 = {}, o1 = {}, o2 = {}, o3 = {};
  float mrow[8], lrow[8];
#pragma unroll
  for (int j = 0; j < 8; ++j) { mrow[j] = -1e30f; lrow[j] = 0.f; }

  __hip_bfloat16* st = stage[warp];

  for (int kb = 0; kb < SEQ; kb += 32) {
    // S tiles: cols kb..kb+15 (s0) and kb+16..kb+31 (s1)
    const __hip_bfloat16* krow0 = Kp + (size_t)(kb + r) * HD + hw * 8;
    const __hip_bfloat16* krow1 = krow0 + 16 * HD;
    v8f s0 = {}, s1 = {};
    s0 = wmma_bf16(aq0, load_frag(krow0), s0);
    s0 = wmma_bf16(aq1, load_frag(krow0 + 32), s0);
    s1 = wmma_bf16(aq0, load_frag(krow1), s1);
    s1 = wmma_bf16(aq1, load_frag(krow1 + 32), s1);

    // online softmax over these 32 columns; C-layout row M = hw*8+j
#pragma unroll
    for (int j = 0; j < 8; ++j) {
      float lm = fmaxf(s0[j], s1[j]);
      lm = fmaxf(lm, __shfl_xor(lm, 1));
      lm = fmaxf(lm, __shfl_xor(lm, 2));
      lm = fmaxf(lm, __shfl_xor(lm, 4));
      lm = fmaxf(lm, __shfl_xor(lm, 8));
      const float mnew = fmaxf(mrow[j], lm);
      const float corr = __expf(mrow[j] - mnew);
      const float p0 = __expf(s0[j] - mnew);
      const float p1 = __expf(s1[j] - mnew);
      float ps = p0 + p1;
      ps += __shfl_xor(ps, 1);
      ps += __shfl_xor(ps, 2);
      ps += __shfl_xor(ps, 4);
      ps += __shfl_xor(ps, 8);
      lrow[j] = lrow[j] * corr + ps;
      mrow[j] = mnew;
      o0[j] *= corr; o1[j] *= corr; o2[j] *= corr; o3[j] *= corr;
      const int row = hw * 8 + j;
      st[row * 32 + r]      = __float2bfloat16(p0);
      st[row * 32 + 16 + r] = __float2bfloat16(p1);
    }

    // re-layout P (C-layout -> A-layout) via per-wave LDS staging
    asm volatile("s_wait_dscnt 0" ::: "memory");
    v16bf pf;
    {
      union { v16bf v; uint4 q[2]; } u;
      const __hip_bfloat16* pp = st + r * 32 + hw * 8;
      u.q[0] = *(const uint4*)pp;
      u.q[1] = *(const uint4*)(pp + 16);
      pf = u.v;
    }

    // O += P(16x32) @ V(32x64): 4 WMMAs over d-chunks; Vt is [d][n] -> contiguous
#pragma unroll
    for (int t = 0; t < 4; ++t) {
      const __hip_bfloat16* vrow = Vp + (size_t)(t * 16 + r) * SEQ + kb + hw * 8;
      const v16bf vf = load_frag(vrow);
      v8f& ot = (t == 0) ? o0 : (t == 1) ? o1 : (t == 2) ? o2 : o3;
      ot = wmma_bf16(pf, vf, ot);
    }
  }

  // finalize: divide by row sums, store bf16 [B*N][C] for the proj GEMM
#pragma unroll
  for (int j = 0; j < 8; ++j) {
    const float inv = 1.f / lrow[j];
    const int nrow = qb * 16 + hw * 8 + j;
    const size_t base = ((size_t)b * SEQ + nrow) * CIN + head * HD;
    Aout[base + 0 * 16 + r] = __float2bfloat16(o0[j] * inv);
    Aout[base + 1 * 16 + r] = __float2bfloat16(o1[j] * inv);
    Aout[base + 2 * 16 + r] = __float2bfloat16(o2[j] * inv);
    Aout[base + 3 * 16 + r] = __float2bfloat16(o3[j] * inv);
  }
}

// -------------------- stage 4: output projection ---------------------------
// 32x64 macro-tile per wave (2x4) for wider spread over the WGPs.
__global__ __launch_bounds__(256)
void proj_gemm_kernel(const __hip_bfloat16* __restrict__ Aout,
                      const __hip_bfloat16* __restrict__ wT,
                      const float* __restrict__ bias,
                      float* __restrict__ out) {
  const int wave = (blockIdx.x * blockDim.x + threadIdx.x) >> 5;
  const int tn = wave % (CIN / 64);     // 16 N macro-tiles
  const int tm = wave / (CIN / 64);     // 128 M macro-tiles
  const int m0 = tm * 32, n0 = tn * 64;

  v8f acc[2][4];
#pragma unroll
  for (int mi = 0; mi < 2; ++mi)
#pragma unroll
    for (int ni = 0; ni < 4; ++ni) acc[mi][ni] = (v8f){};
  wave_gemm_block<2, 4>(Aout, wT, CIN, CIN, m0, n0, CIN, acc);

  const int lane = threadIdx.x & 31;
  const int r = lane & 15, hw = lane >> 4;
#pragma unroll
  for (int ni = 0; ni < 4; ++ni) {
    const int col = n0 + ni * 16 + r;
    const float bv = bias[col];
#pragma unroll
    for (int mi = 0; mi < 2; ++mi) {
#pragma unroll
      for (int j = 0; j < 8; ++j) {
        const int m = m0 + mi * 16 + hw * 8 + j;
        out[(size_t)m * CIN + col] = acc[mi][ni][j] + bv;
      }
    }
  }
}

// -------------------- host launch ------------------------------------------

extern "C" void kernel_launch(void* const* d_in, const int* in_sizes, int n_in,
                              void* d_out, int out_size, void* d_ws, size_t ws_size,
                              hipStream_t stream) {
  const float* x      = (const float*)d_in[0];
  const float* w_qkv  = (const float*)d_in[1];
  const float* b_qkv  = (const float*)d_in[2];
  const float* w_proj = (const float*)d_in[3];
  const float* b_proj = (const float*)d_in[4];
  float* out = (float*)d_out;

  char* ws = (char*)d_ws;
  size_t off = 0;
  auto alloc = [&](size_t bytes) -> void* {
    void* p = ws + off;
    off = (off + bytes + 255) & ~(size_t)255;
    return p;
  };
  __hip_bfloat16* xb     = (__hip_bfloat16*)alloc((size_t)BN * CIN * 2);   // 8 MB
  __hip_bfloat16* wqkvT  = (__hip_bfloat16*)alloc((size_t)C3 * CIN * 2);   // 6 MB
  __hip_bfloat16* wprojT = (__hip_bfloat16*)alloc((size_t)CIN * CIN * 2);  // 2 MB
  __hip_bfloat16* Qb     = (__hip_bfloat16*)alloc((size_t)NBH * SEQ * HD * 2); // 8 MB
  __hip_bfloat16* Kb     = (__hip_bfloat16*)alloc((size_t)NBH * SEQ * HD * 2); // 8 MB
  __hip_bfloat16* Vt     = (__hip_bfloat16*)alloc((size_t)NBH * HD * SEQ * 2); // 8 MB
  __hip_bfloat16* Aout   = (__hip_bfloat16*)alloc((size_t)BN * CIN * 2);   // 8 MB
  (void)ws_size; (void)in_sizes; (void)n_in; (void)out_size;

  // stage 1: conversions / transposes
  cvt_f32_bf16<<<(BN * CIN + 255) / 256, 256, 0, stream>>>(x, xb, BN * CIN);
  transpose_cvt<<<(CIN * C3 + 255) / 256, 256, 0, stream>>>(w_qkv, wqkvT, CIN, C3);
  transpose_cvt<<<(CIN * CIN + 255) / 256, 256, 0, stream>>>(w_proj, wprojT, CIN, CIN);

  // stage 2: QKV projection (64x48 macro-tiles, 8 waves/block)
  qkv_gemm_kernel<<<(BN / 64) * (C3 / 64) / 8, 256, 0, stream>>>(
      xb, wqkvT, b_qkv, Qb, Kb, Vt);

  // stage 3: attention (32 bh * 128 qblocks = 4096 wave tasks)
  attn_kernel<<<NBH * (SEQ / 16) / 8, 256, 0, stream>>>(Qb, Kb, Vt, Aout);

  // stage 4: output projection (128x16 macro-tiles of 32x64)
  proj_gemm_kernel<<<(BN / 32) * (CIN / 64) / 8, 256, 0, stream>>>(
      Aout, wprojT, b_proj, out);
}
